// AgentBASELINE_13915694039393
// MI455X (gfx1250) — matrix operand order repaired and checked by reading, not compiled, not run on hardware
//
#include <hip/hip_runtime.h>

// ---------------------------------------------------------------------------
// Fully-fused MLP for MI455X (gfx1250), bf16 WMMA path.
//   B=131072 rows, H=512.  5 x (512x512) GEMMs on v_wmma_f32_16x16x32_bf16,
//   everything else fused. Activations live in LDS; weights prepacked to bf16
//   fragment-major layout in d_ws (read from L2 by every workgroup).
// ---------------------------------------------------------------------------

typedef __bf16 bf16_t;
typedef bf16_t v16bf __attribute__((ext_vector_type(16)));
typedef bf16_t v8bf  __attribute__((ext_vector_type(8)));
typedef float  v8f   __attribute__((ext_vector_type(8)));

#define BTOT   131072
#define HDIM   512
#define MT     64            // batch rows per workgroup
#define LSTR   520           // LDS activation row stride (bf16 elems), padded
#define HSTR   68            // LDS head-out row stride (f32), padded

// packed weight offsets in d_ws (bf16 element units)
#define PK_MAT_ELEMS (512*512)
#define PK_F0W2 (0*PK_MAT_ELEMS)
#define PK_F0W3 (1*PK_MAT_ELEMS)
#define PK_F2W2 (2*PK_MAT_ELEMS)
#define PK_F2W3 (3*PK_MAT_ELEMS)
#define PK_F2W4 (4*PK_MAT_ELEMS)
#define PK_HEAD (5*PK_MAT_ELEMS)
#define PK_TOTAL (5*PK_MAT_ELEMS + 512*64)   // 1,343,488 bf16 = ~2.6 MB

// ---------------------------------------------------------------------------
// Weight prepack: f32 -> bf16, fragment-major layout matching the CDNA5
// 16-bit B-matrix (32x16) per-lane VGPR layout:
//   element e of lane L (n16=L&15, h=L>>4) holds W[K = kt*32 + (e&7) + 8*h + 16*(e>>3)][N = nt*16 + n16]
// Packed index: ((nt*16 + kt)*32 + lane)*16 + e  -> each lane's fragment is a
// contiguous 32B chunk, loadable with two global_load_b128.
// ---------------------------------------------------------------------------
__global__ __launch_bounds__(256) void cdna5_prepack(
    const float* __restrict__ f0w2, const float* __restrict__ f0w3,
    const float* __restrict__ f2w2, const float* __restrict__ f2w3,
    const float* __restrict__ f2w4, const float* __restrict__ f4w,
    const float* __restrict__ f5w,  const float* __restrict__ f6w,
    const float* __restrict__ f7w,  bf16_t* __restrict__ pk)
{
  int idx = blockIdx.x * 256 + threadIdx.x;
  if (idx >= PK_TOTAL) return;
  float v;
  if (idx < PK_HEAD) {
    int mat = idx / PK_MAT_ELEMS;
    int r   = idx - mat * PK_MAT_ELEMS;
    int ntile = r >> 13;            // 16*32*16 = 8192 elems per n-tile
    int rem   = r & 8191;
    int kt    = rem >> 9;           // 32*16 = 512 elems per k-tile
    int lane  = (rem >> 4) & 31;
    int e     = rem & 15;
    int K = kt * 32 + (e & 7) + 8 * (lane >> 4) + 16 * (e >> 3);
    int N = ntile * 16 + (lane & 15);
    const float* W = (mat == 0) ? f0w2 : (mat == 1) ? f0w3
                   : (mat == 2) ? f2w2 : (mat == 3) ? f2w3 : f2w4;
    v = W[K * 512 + N];
  } else {
    int r = idx - PK_HEAD;          // heads: [512, 64] = f4w|f5w|f6w|f7w|pad
    int ntile = r >> 13;
    int rem   = r & 8191;
    int kt    = rem >> 9;
    int lane  = (rem >> 4) & 31;
    int e     = rem & 15;
    int K = kt * 32 + (e & 7) + 8 * (lane >> 4) + 16 * (e >> 3);
    int N = ntile * 16 + (lane & 15);
    if      (N < 16) v = f4w[K * 16 + N];
    else if (N < 48) v = f5w[K * 32 + (N - 16)];
    else if (N < 56) v = f6w[K * 8  + (N - 48)];
    else if (N == 56) v = f7w[K];
    else v = 0.0f;
  }
  pk[idx] = (bf16_t)v;
}

// ---------------------------------------------------------------------------
// Fragment loaders.
// A (16x32 bf16, from LDS): lane (m16=L&15, h=L>>4), elems 0..7 -> K=kk+8h..,
// elems 8..15 -> K=kk+16+8h.. : two 16B LDS reads per lane.
// ---------------------------------------------------------------------------
__device__ __forceinline__ v16bf load_a_frag(const bf16_t* act, int mtile, int kt, int lane)
{
  const int m16 = lane & 15, hh = lane >> 4;
  const bf16_t* p = act + (mtile * 16 + m16) * LSTR + kt * 32 + 8 * hh;
  v8bf lo = *(const v8bf*)p;
  v8bf hi = *(const v8bf*)(p + 16);
  v16bf av;
#pragma unroll
  for (int i = 0; i < 8; ++i) { av[i] = lo[i]; av[i + 8] = hi[i]; }
  return av;
}

__device__ __forceinline__ v16bf load_b_frag(const bf16_t* wfrag, int ntile, int kt, int lane)
{
  // contiguous 32B per lane (prepacked) -> compiler emits 2x global_load_b128
  return *(const v16bf*)(wfrag + ((((ntile << 4) + kt) << 5) + lane) * 16);
}

// ---------------------------------------------------------------------------
// WMMA GEMM layer over the 64-row tile.
//  MODE 0: out = relu(in @ W + bias)                 (N=512, bf16 out to LDS)
//  MODE 1: out = relu(in @ W + f1b + s0 @ f1w)      (N=512, bf16 out to LDS)
//  MODE 2: headOut = in @ Whead                      (N=64,  f32  out to LDS)
// Wave w: m-tile = w&3 ; n-tiles split by w>>2. Uniform control flow so EXEC
// is all-ones around every v_wmma (ISA requirement).
// ---------------------------------------------------------------------------
template <int MODE>
__device__ void wmma_layer(const bf16_t* __restrict__ actIn, bf16_t* __restrict__ actOut,
                           const bf16_t* __restrict__ wfrag,
                           const float* __restrict__ bias,
                           const float* __restrict__ f1w, const float* __restrict__ f1b,
                           const float* __restrict__ s0T, float* __restrict__ hOut,
                           int wave, int lane)
{
  const int m    = wave & 3;
  const int half = wave >> 2;
  const int n16  = lane & 15;
  const int hh   = lane >> 4;
  const v8f z = {0.f, 0.f, 0.f, 0.f, 0.f, 0.f, 0.f, 0.f};

  if (MODE == 2) {
    const int nbase = half * 2;               // 4x4 tiles, 2 per wave
    v8f acc[2] = {z, z};
    for (int kt = 0; kt < 16; ++kt) {
      v16bf av = load_a_frag(actIn, m, kt, lane);
#pragma unroll
      for (int j = 0; j < 2; ++j) {
        v16bf bv = load_b_frag(wfrag, nbase + j, kt, lane);
        acc[j] = __builtin_amdgcn_wmma_f32_16x16x32_bf16(
            false, av, false, bv, (short)0, acc[j], false, false);
      }
    }
#pragma unroll
    for (int j = 0; j < 2; ++j) {
      int N = (nbase + j) * 16 + n16;
#pragma unroll
      for (int g = 0; g < 8; ++g) {
        int M = m * 16 + g + 8 * hh;
        hOut[M * HSTR + N] = acc[j][g];
      }
    }
  } else {
    for (int ng = 0; ng < 16; ng += 4) {      // 16 n-tiles per wave, 4 at a time
      v8f acc[4] = {z, z, z, z};
      for (int kt = 0; kt < 16; ++kt) {
        v16bf av = load_a_frag(actIn, m, kt, lane);   // reused by 4 WMMAs
#pragma unroll
        for (int j = 0; j < 4; ++j) {
          v16bf bv = load_b_frag(wfrag, half * 16 + ng + j, kt, lane);
          acc[j] = __builtin_amdgcn_wmma_f32_16x16x32_bf16(
              false, av, false, bv, (short)0, acc[j], false, false);
        }
      }
#pragma unroll
      for (int j = 0; j < 4; ++j) {
        int N = (half * 16 + ng + j) * 16 + n16;
        float bN, w0 = 0.f, w1 = 0.f, w2 = 0.f, w3 = 0.f;
        if (MODE == 0) {
          bN = bias[N];
        } else {
          bN = f1b[N];
          w0 = f1w[N]; w1 = f1w[512 + N]; w2 = f1w[1024 + N]; w3 = f1w[1536 + N];
        }
#pragma unroll
        for (int g = 0; g < 8; ++g) {
          int M = m * 16 + g + 8 * hh;
          float v = acc[j][g] + bN;
          if (MODE == 1)
            v += s0T[M * 4 + 0] * w0 + s0T[M * 4 + 1] * w1 +
                 s0T[M * 4 + 2] * w2 + s0T[M * 4 + 3] * w3;
          actOut[M * LSTR + N] = (bf16_t)(v > 0.f ? v : 0.f);
        }
      }
    }
  }
}

// K=4 input layer: out[64,512] = relu(in @ W[4,512] + b), scalar f32, bf16 out
__device__ __forceinline__ void dense4(const float* __restrict__ inT,
                                       const float* __restrict__ W,
                                       const float* __restrict__ b,
                                       bf16_t* __restrict__ actOut, int tid)
{
#pragma unroll 4
  for (int it = 0; it < (MT * HDIM) / 256; ++it) {
    int idx = it * 256 + tid;
    int r = idx >> 9, n = idx & 511;
    float x0 = inT[r * 4 + 0], x1 = inT[r * 4 + 1];
    float x2 = inT[r * 4 + 2], x3 = inT[r * 4 + 3];
    float v = b[n] + x0 * W[n] + x1 * W[512 + n] + x2 * W[1024 + n] + x3 * W[1536 + n];
    actOut[r * LSTR + n] = (bf16_t)(v > 0.f ? v : 0.f);
  }
}

// ---------------------------------------------------------------------------
// Main fused kernel: one workgroup = 64 batch rows, 8 wave32.
// ---------------------------------------------------------------------------
__global__ __launch_bounds__(256, 1) void cdna5_mlp_fused(
    const float* __restrict__ s,    const float* __restrict__ a,
    const float* __restrict__ f0w1, const float* __restrict__ f0b1,
    const float* __restrict__ f0b2, const float* __restrict__ f0b3,
    const float* __restrict__ f0w4, const float* __restrict__ f0b4,
    const float* __restrict__ f1w,  const float* __restrict__ f1b,
    const float* __restrict__ f2w1, const float* __restrict__ f2b1,
    const float* __restrict__ f2b2, const float* __restrict__ f2b3,
    const bf16_t* __restrict__ pk,  float* __restrict__ out)
{
  extern __shared__ char smem[];
  bf16_t* actA = (bf16_t*)smem;                  // 64*520*2 = 66,560 B
  bf16_t* actB = actA + MT * LSTR;               // 66,560 B
  float*  sT   = (float*)(actB + MT * LSTR);     // 64*4 f32
  float*  s0T  = sT + MT * 4;                    // 64*4 f32
  float*  aT   = s0T + MT * 4;                   // 64*2 f32
  float*  hOut = aT + MT * 2;                    // 64*68 f32 = 17,408 B
  // total ~153 KB of the 320 KB WGP LDS

  const int tid  = threadIdx.x;
  const int wave = tid >> 5;
  const int lane = tid & 31;
  const int R    = blockIdx.x * MT;

  // stage s, a tiles
  if (tid < MT * 4) sT[tid] = s[R * 4 + tid];
  if (tid < MT * 2) aT[tid] = a[R * 2 + tid];
  __syncthreads();

  // ---- fm0 ----
  dense4(sT, f0w1, f0b1, actA, tid);
  __syncthreads();
  wmma_layer<0>(actA, actB, pk + PK_F0W2, f0b2, nullptr, nullptr, nullptr, nullptr, wave, lane);
  __syncthreads();
  wmma_layer<0>(actB, actA, pk + PK_F0W3, f0b3, nullptr, nullptr, nullptr, nullptr, wave, lane);
  __syncthreads();

  // s_ = h @ f0w4 + f0b4 ; s0 = s - s_   (256 threads = one output each)
  {
    int r = tid >> 2, c = tid & 3;
    float accv = f0b4[c];
    const bf16_t* row = actA + r * LSTR;
#pragma unroll 8
    for (int k = 0; k < HDIM; ++k) accv += (float)row[k] * f0w4[k * 4 + c];
    s0T[r * 4 + c] = sT[r * 4 + c] - accv;
  }
  __syncthreads();

  // ---- fm2, fused with fm1 skip-path in the last layer's epilogue ----
  dense4(s0T, f2w1, f2b1, actA, tid);
  __syncthreads();
  wmma_layer<0>(actA, actB, pk + PK_F2W2, f2b2, nullptr, nullptr, nullptr, nullptr, wave, lane);
  __syncthreads();
  wmma_layer<0>(actB, actA, pk + PK_F2W3, f2b3, nullptr, nullptr, nullptr, nullptr, wave, lane);
  __syncthreads();
  // hid = relu(g @ f2w4 + s0 @ f1w + f1b)  -> actB (bf16)
  wmma_layer<1>(actA, actB, pk + PK_F2W4, nullptr, f1w, f1b, s0T, nullptr, wave, lane);
  __syncthreads();

  // ---- heads: hid @ [f4w|f5w|f6w|f7w|pad]  -> hOut[64,64] f32 ----
  wmma_layer<2>(actB, nullptr, pk + PK_HEAD, nullptr, nullptr, nullptr, nullptr, hOut, wave, lane);
  __syncthreads();

  // ---- per-row J/mean contraction (note s0 + s_ == s) ----
  if (tid < MT) {
    const float* hv = hOut + tid * HSTR;   // [0,16)=A  [16,48)=Bt  [48,56)=C  56=o
    float a0 = aT[tid * 2 + 0], a1 = aT[tid * 2 + 1];
    float o  = hv[56];
#pragma unroll
    for (int j = 0; j < 4; ++j) {
      float mval = o + a0 * hv[48 + j] + a1 * hv[48 + 4 + j];
#pragma unroll
      for (int i = 0; i < 4; ++i) {
        int ij = i * 4 + j;
        float J = hv[ij] + a0 * hv[16 + 2 * ij] + a1 * hv[16 + 2 * ij + 1];
        mval += sT[tid * 4 + i] * J;
      }
      out[(R + tid) * 4 + j] = mval;
    }
  }
}

// ---------------------------------------------------------------------------
extern "C" void kernel_launch(void* const* d_in, const int* in_sizes, int n_in,
                              void* d_out, int out_size, void* d_ws, size_t ws_size,
                              hipStream_t stream)
{
  (void)in_sizes; (void)n_in; (void)out_size; (void)ws_size;
  const float* s    = (const float*)d_in[0];
  const float* a    = (const float*)d_in[1];
  const float* f0w1 = (const float*)d_in[2];
  const float* f0b1 = (const float*)d_in[3];
  const float* f0w2 = (const float*)d_in[4];
  const float* f0b2 = (const float*)d_in[5];
  const float* f0w3 = (const float*)d_in[6];
  const float* f0b3 = (const float*)d_in[7];
  const float* f0w4 = (const float*)d_in[8];
  const float* f0b4 = (const float*)d_in[9];
  const float* f1w  = (const float*)d_in[10];
  const float* f1b  = (const float*)d_in[11];
  const float* f2w1 = (const float*)d_in[12];
  const float* f2b1 = (const float*)d_in[13];
  const float* f2w2 = (const float*)d_in[14];
  const float* f2b2 = (const float*)d_in[15];
  const float* f2w3 = (const float*)d_in[16];
  const float* f2b3 = (const float*)d_in[17];
  const float* f2w4 = (const float*)d_in[18];
  const float* f4w  = (const float*)d_in[19];
  const float* f5w  = (const float*)d_in[20];
  const float* f6w  = (const float*)d_in[21];
  const float* f7w  = (const float*)d_in[22];
  bf16_t* pk = (bf16_t*)d_ws;    // needs ~2.6 MB of scratch

  cdna5_prepack<<<(PK_TOTAL + 255) / 256, 256, 0, stream>>>(
      f0w2, f0w3, f2w2, f2w3, f2w4, f4w, f5w, f6w, f7w, pk);

  const size_t smem_bytes =
      (size_t)(2 * MT * LSTR) * sizeof(bf16_t) +
      (size_t)(MT * 4 + MT * 4 + MT * 2 + MT * HSTR) * sizeof(float);

  cdna5_mlp_fused<<<BTOT / MT, 256, smem_bytes, stream>>>(
      s, a, f0w1, f0b1, f0b2, f0b3, f0w4, f0b4,
      f1w, f1b, f2w1, f2b1, f2b2, f2b3, pk, (float*)d_out);
}
